// HodgeTorsionAttention_15298673508558
// MI455X (gfx1250) — matrix operand order repaired and checked by reading, not compiled, not run on hardware
//
#include <hip/hip_runtime.h>
#include <hip/hip_bf16.h>
#include <stdint.h>

typedef __bf16 bf16;
typedef __attribute__((ext_vector_type(16))) __bf16 v16bf;
typedef __attribute__((ext_vector_type(8)))  __bf16 v8bf;
typedef __attribute__((ext_vector_type(8)))  float  v8f;

#define WMMA_BF16(a, b, c) \
  __builtin_amdgcn_wmma_f32_16x16x32_bf16(false, (a), false, (b), (short)0, (c), false, false)

// Problem constants (fixed shapes from the reference)
constexpr int BATCH = 2;
constexpr int SEQ   = 2048;
constexpr int DMODEL= 1024;
constexpr int NHEAD = 16;
constexpr int HDIM  = 64;
constexpr int MROWS = BATCH * SEQ;        // 4096
__device__ constexpr float PHI_F  = 1.6180339887f;
__device__ constexpr float GIZA_F = 0.9049532172f; // 51.85 * pi / 180

// ---------------------------------------------------------------------------
// Fragment loaders (wave32)
// A-fragment 16x32 (MxK), row-major source with leading dim ld (elements):
//   lane: row = lane&15 ; e<8 -> k = 8*half + e ; e>=8 -> k = 16 + 8*half + (e-8)
__device__ __forceinline__ v16bf load_a_frag(const bf16* p, int ld, int lane) {
  int hf = lane >> 4, r = lane & 15;
  const bf16* base = p + r * ld + 8 * hf;
  v8bf lo = *(const v8bf*)(base);
  v8bf hi = *(const v8bf*)(base + 16);
  return __builtin_shufflevector(lo, hi, 0,1,2,3,4,5,6,7,8,9,10,11,12,13,14,15);
}
// B-fragment 32x16 (KxN), columns stored contiguously (source is B^T row-major):
//   lane: col = lane&15 ; e<8 -> k = 8*half + e ; e>=8 -> k = 16 + 8*half + (e-8)
__device__ __forceinline__ v16bf load_b_frag(const bf16* p, int ld, int lane) {
  int hf = lane >> 4, n = lane & 15;
  const bf16* base = p + n * ld + 8 * hf;
  v8bf lo = *(const v8bf*)(base);
  v8bf hi = *(const v8bf*)(base + 16);
  return __builtin_shufflevector(lo, hi, 0,1,2,3,4,5,6,7,8,9,10,11,12,13,14,15);
}

// Row reductions across the 16 lanes holding one C-matrix row.
__device__ __forceinline__ float rowmax16(float v) {
#pragma unroll
  for (int m = 8; m >= 1; m >>= 1) v = fmaxf(v, __shfl_xor(v, m, 32));
  return v;
}
__device__ __forceinline__ float rowsum16(float v) {
#pragma unroll
  for (int m = 8; m >= 1; m >>= 1) v += __shfl_xor(v, m, 32);
  return v;
}

// ---------------------------------------------------------------------------
// Prep kernels
__global__ void cvt_bf16_kernel(const float* __restrict__ in, bf16* __restrict__ out, int n) {
  int i = blockIdx.x * blockDim.x + threadIdx.x;
  if (i < n) out[i] = (bf16)in[i];
}

// Coalesced tiled transpose: out[w][n][k] = (bf16) W_w[k][n]
__global__ __launch_bounds__(256) void transpose_w_kernel(
    const float* __restrict__ Wq, const float* __restrict__ Wk,
    const float* __restrict__ Wv, const float* __restrict__ Wo,
    bf16* __restrict__ out) {
  __shared__ bf16 tile[32][33];
  int w = blockIdx.z;
  const float* W = (w == 0) ? Wq : (w == 1) ? Wk : (w == 2) ? Wv : Wo;
  int n0 = blockIdx.x * 32;   // column block of W (= row block of out)
  int k0 = blockIdx.y * 32;   // row block of W
  int tx = threadIdx.x & 31, ty = threadIdx.x >> 5;  // 32 x 8
#pragma unroll
  for (int i = 0; i < 4; ++i)
    tile[ty + i * 8][tx] = (bf16)W[(size_t)(k0 + ty + i * 8) * DMODEL + (n0 + tx)];
  __syncthreads();
  size_t obase = ((size_t)w << 20);
#pragma unroll
  for (int i = 0; i < 4; ++i)
    out[obase + (size_t)(n0 + ty + i * 8) * DMODEL + (k0 + tx)] = tile[tx][ty + i * 8];
}

__global__ void torsion_kernel(float* __restrict__ tor) {
  int i = blockIdx.x * blockDim.x + threadIdx.x;
  if (i < 2 * SEQ - 1) {
    float rel = (float)(i - (SEQ - 1));          // rel = j - i
    tor[i] = PHI_F * sinf(GIZA_F * rel);
  }
}

// ---------------------------------------------------------------------------
// WMMA GEMM with async-to-LDS B staging.
// Block = 256 threads (8 waves) computes a 128x64 output tile; the 64-column
// B panel is staged into LDS by GLOBAL_LOAD_ASYNC_TO_LDS_B128, double-buffered
// in 32-deep K chunks (copy of chunk k+1 overlaps WMMA on chunk k).
// mode: 0 = f32 row-major, 1 = bf16 [b,h,s,hd] (*scale), 2 = bf16 [b,h,hd,s].
constexpr int BROW_PAD = 40;                      // LDS row stride (elems): 80B = 20 banks
constexpr int BBUF_ELEMS = 64 * BROW_PAD;         // one K-chunk panel

// Issue one per-lane 16B async copy global->LDS. The LDS address operand is
// derived from the real LDS pointer so the shared object escapes into the asm
// (otherwise alias analysis deletes the ds reads of the staged data).
__device__ __forceinline__ void async_copy_b128(bf16* lds_ptr, const bf16* gptr) {
  unsigned loff = (unsigned)(uintptr_t)lds_ptr;   // low 32 bits = LDS byte offset
  asm volatile("global_load_async_to_lds_b128 %0, %1, off"
               :: "v"(loff), "v"(gptr) : "memory");
}

__global__ __launch_bounds__(256) void gemm_wmma_kernel(
    const bf16* __restrict__ A, const bf16* __restrict__ Bt,
    const float* __restrict__ bias,
    float* __restrict__ outF, bf16* __restrict__ outH,
    int mode, float scale) {
  __shared__ bf16 bs[2 * BBUF_ELEMS];             // 2 x 5120 B double buffer

  int tid  = threadIdx.x;
  int lane = tid & 31;
  int wv   = tid >> 5;                            // wave 0..7
  int m0 = blockIdx.x * 128 + wv * 16;            // 32 row blocks
  int n0 = blockIdx.y * 64;                       // 16 column blocks

  // async staging assignment: each thread copies one 16B segment per chunk
  int srow = tid >> 2;                            // 0..63 (column n within panel)
  int sseg = tid & 3;                             // 0..3 (16B = 8 bf16)
  const bf16* gsrc0 = Bt + (size_t)(n0 + srow) * DMODEL + sseg * 8;
  bf16* lds0 = &bs[srow * BROW_PAD + sseg * 8];
  bf16* lds1 = lds0 + BBUF_ELEMS;
  const bf16* arow = A + (size_t)m0 * DMODEL;

  v8f acc[4] = {};

  // one pipeline phase: consume chunk kc from `bp`, prefetch chunk kc+1 into `nxt`
  auto phase = [&](int kc, const bf16* bp, bf16* nxt) {
    asm volatile("s_wait_asynccnt 0x0" ::: "memory");   // my chunk-kc copy landed
    __syncthreads();                                    // everyone's chunk kc visible
    if (kc + 1 < DMODEL / 32)                           // overlap: copy chunk kc+1
      async_copy_b128(nxt, gsrc0 + (kc + 1) * 32);
    v16bf a  = load_a_frag(arow + kc * 32, DMODEL, lane);
    v16bf b0 = load_b_frag(bp + 0 * 16 * BROW_PAD, BROW_PAD, lane);
    v16bf b1 = load_b_frag(bp + 1 * 16 * BROW_PAD, BROW_PAD, lane);
    v16bf b2 = load_b_frag(bp + 2 * 16 * BROW_PAD, BROW_PAD, lane);
    v16bf b3 = load_b_frag(bp + 3 * 16 * BROW_PAD, BROW_PAD, lane);
    acc[0] = WMMA_BF16(a, b0, acc[0]);
    acc[1] = WMMA_BF16(a, b1, acc[1]);
    acc[2] = WMMA_BF16(a, b2, acc[2]);
    acc[3] = WMMA_BF16(a, b3, acc[3]);
  };

  async_copy_b128(lds0, gsrc0);                   // chunk 0
  const bf16* rb0 = &bs[0];
  const bf16* rb1 = &bs[BBUF_ELEMS];
#pragma unroll 1
  for (int kc = 0; kc < DMODEL / 32; kc += 2) {   // explicit two-phase double buffer
    phase(kc,     rb0, lds1);
    phase(kc + 1, rb1, lds0);
  }

  int hf = lane >> 4, nl = lane & 15;
#pragma unroll
  for (int t = 0; t < 4; ++t) {
    int n = n0 + t * 16 + nl;
    float bn = bias[n];
#pragma unroll
    for (int g = 0; g < 8; ++g) {
      int m = m0 + g + 8 * hf;
      float v = acc[t][g] + bn;
      if (mode == 0) {
        outF[(size_t)m * DMODEL + n] = v;
      } else if (mode == 1) {            // [b, h, s, hd]
        int b_ = m >> 11, s = m & (SEQ - 1);
        int h = n >> 6, d = n & (HDIM - 1);
        outH[((((size_t)b_ * NHEAD + h) * SEQ + s) << 6) + d] = (bf16)(v * scale);
      } else {                           // [b, h, hd, s]
        int b_ = m >> 11, s = m & (SEQ - 1);
        int h = n >> 6, d = n & (HDIM - 1);
        outH[(((size_t)b_ * NHEAD + h) * HDIM + d) * SEQ + s] = (bf16)v;
      }
    }
  }
}

// ---------------------------------------------------------------------------
// Flash attention with torsion bias. One wave per (b,h) x 16-query tile,
// streaming keys in blocks of 32 with online softmax. V fragments are loaded
// before the softmax VALU block so their latency is hidden.
__global__ __launch_bounds__(256) void flash_torsion_kernel(
    const bf16* __restrict__ Qp,   // [b,h,s,hd] bf16, pre-scaled by 1/sqrt(hd)
    const bf16* __restrict__ Kp,   // [b,h,s,hd] bf16
    const bf16* __restrict__ Vt,   // [b,h,hd,s] bf16
    const float* __restrict__ tor, // [2S-1], index = (j - i) + S-1
    bf16* __restrict__ attn) {     // [b,s,D] bf16
  __shared__ bf16 pbuf[8][16 * 32];               // per-wave P staging

  int lane  = threadIdx.x & 31;
  int wslot = threadIdx.x >> 5;
  int wid   = (blockIdx.x * blockDim.x + threadIdx.x) >> 5;  // 0..4095
  int bh = wid >> 7;
  int q0 = (wid & 127) * 16;
  int hf = lane >> 4, nl = lane & 15;

  const bf16* Qb = Qp + (size_t)bh * SEQ * HDIM;
  const bf16* Kb = Kp + (size_t)bh * SEQ * HDIM;
  const bf16* Vb = Vt + (size_t)bh * HDIM * SEQ;
  bf16* myp = &pbuf[wslot][0];

  v16bf qa0 = load_a_frag(Qb + q0 * HDIM + 0,  HDIM, lane);
  v16bf qa1 = load_a_frag(Qb + q0 * HDIM + 32, HDIM, lane);

  v8f o0 = {}, o1 = {}, o2 = {}, o3 = {};
  float mrow[8], lrow[8];
#pragma unroll
  for (int g = 0; g < 8; ++g) { mrow[g] = -3.0e38f; lrow[g] = 0.f; }

  for (int j = 0; j < SEQ; j += 32) {
    // ---- scores for 32 keys: two 16x16 tiles ----
    v8f s0 = {}, s1 = {};
    {
      v16bf b0 = load_b_frag(Kb + (size_t)j * HDIM + 0,  HDIM, lane);
      s0 = WMMA_BF16(qa0, b0, s0);
      v16bf b1 = load_b_frag(Kb + (size_t)j * HDIM + 32, HDIM, lane);
      s0 = WMMA_BF16(qa1, b1, s0);
      v16bf b2 = load_b_frag(Kb + (size_t)(j + 16) * HDIM + 0,  HDIM, lane);
      s1 = WMMA_BF16(qa0, b2, s1);
      v16bf b3 = load_b_frag(Kb + (size_t)(j + 16) * HDIM + 32, HDIM, lane);
      s1 = WMMA_BF16(qa1, b3, s1);
    }
    // issue V fragment loads now; latency hidden behind softmax VALU
    v16bf vb0 = load_b_frag(Vb + 0  * SEQ + j, SEQ, lane);
    v16bf vb1 = load_b_frag(Vb + 16 * SEQ + j, SEQ, lane);
    v16bf vb2 = load_b_frag(Vb + 32 * SEQ + j, SEQ, lane);
    v16bf vb3 = load_b_frag(Vb + 48 * SEQ + j, SEQ, lane);

    // ---- torsion bias: element (m, n) -> tor[(j+n) - (q0+m) + S-1] ----
    int d0 = j + nl - (q0 + 8 * hf) + (SEQ - 1);
#pragma unroll
    for (int g = 0; g < 8; ++g) {
      s0[g] += tor[d0 - g];
      s1[g] += tor[d0 + 16 - g];
    }
    // ---- online softmax (per output row g) ----
#pragma unroll
    for (int g = 0; g < 8; ++g) {
      float mx = rowmax16(fmaxf(s0[g], s1[g]));
      float mnew = fmaxf(mrow[g], mx);
      float corr = __expf(mrow[g] - mnew);
      float p0 = __expf(s0[g] - mnew);
      float p1 = __expf(s1[g] - mnew);
      float rs = rowsum16(p0 + p1);
      lrow[g] = lrow[g] * corr + rs;
      mrow[g] = mnew;
      o0[g] *= corr; o1[g] *= corr; o2[g] *= corr; o3[g] *= corr;
      int m = g + 8 * hf;
      myp[m * 32 + nl]      = (bf16)p0;
      myp[m * 32 + 16 + nl] = (bf16)p1;
    }
    asm volatile("s_wait_dscnt 0x0" ::: "memory");  // P staged (same-wave LDS dep)
    // ---- O += P[16x32] @ V[32x64] ----
    v16bf pa = load_a_frag(myp, 32, lane);
    o0 = WMMA_BF16(pa, vb0, o0);
    o1 = WMMA_BF16(pa, vb1, o1);
    o2 = WMMA_BF16(pa, vb2, o2);
    o3 = WMMA_BF16(pa, vb3, o3);
  }

  // ---- normalize and store heads-concatenated [b, s, D] ----
  int b_ = bh >> 4, h = bh & 15;
#pragma unroll
  for (int g = 0; g < 8; ++g) {
    int m = q0 + g + 8 * hf;
    float inv = 1.0f / lrow[g];
    size_t rowoff = ((size_t)(b_ * SEQ + m)) * DMODEL + h * HDIM;
    attn[rowoff + 0  + nl] = (bf16)(o0[g] * inv);
    attn[rowoff + 16 + nl] = (bf16)(o1[g] * inv);
    attn[rowoff + 32 + nl] = (bf16)(o2[g] * inv);
    attn[rowoff + 48 + nl] = (bf16)(o3[g] * inv);
  }
}

// ---------------------------------------------------------------------------
extern "C" void kernel_launch(void* const* d_in, const int* in_sizes, int n_in,
                              void* d_out, int out_size, void* d_ws, size_t ws_size,
                              hipStream_t stream) {
  const float* hid = (const float*)d_in[0];
  const float* Wq  = (const float*)d_in[1];
  const float* bq  = (const float*)d_in[2];
  const float* Wk  = (const float*)d_in[3];
  const float* bk  = (const float*)d_in[4];
  const float* Wv  = (const float*)d_in[5];
  const float* bv  = (const float*)d_in[6];
  const float* Wo  = (const float*)d_in[7];
  const float* bo  = (const float*)d_in[8];
  float* out = (float*)d_out;

  char* ws = (char*)d_ws;
  size_t off = 0;
  auto alloc = [&](size_t bytes) -> void* {
    void* p = ws + off;
    off = (off + bytes + 255) & ~(size_t)255;
    return p;
  };
  bf16* hidb = (bf16*)alloc((size_t)MROWS * DMODEL * 2);
  bf16* Wt   = (bf16*)alloc(4ull * DMODEL * DMODEL * 2);   // WqT,WkT,WvT,WoT bf16
  bf16* Qb   = (bf16*)alloc((size_t)MROWS * DMODEL * 2);
  bf16* Kb   = (bf16*)alloc((size_t)MROWS * DMODEL * 2);
  bf16* Vtb  = (bf16*)alloc((size_t)MROWS * DMODEL * 2);
  bf16* attn = (bf16*)alloc((size_t)MROWS * DMODEL * 2);
  float* tor = (float*)alloc((size_t)(2 * SEQ - 1) * 4);

  int nconv = MROWS * DMODEL;
  cvt_bf16_kernel<<<(nconv + 255) / 256, 256, 0, stream>>>(hid, hidb, nconv);
  transpose_w_kernel<<<dim3(32, 32, 4), 256, 0, stream>>>(Wq, Wk, Wv, Wo, Wt);
  torsion_kernel<<<(2 * SEQ - 1 + 255) / 256, 256, 0, stream>>>(tor);

  // Q/K/V projections: grid 32 (M/128) x 16 (N/64) blocks of 256
  dim3 ggrid(32, 16);
  gemm_wmma_kernel<<<ggrid, 256, 0, stream>>>(hidb, Wt,                         bq, nullptr, Qb,  1, 0.125f);
  gemm_wmma_kernel<<<ggrid, 256, 0, stream>>>(hidb, Wt + (size_t)DMODEL*DMODEL, bk, nullptr, Kb,  1, 1.0f);
  gemm_wmma_kernel<<<ggrid, 256, 0, stream>>>(hidb, Wt + 2ull*DMODEL*DMODEL,    bv, nullptr, Vtb, 2, 1.0f);

  // Attention (32 bh * 128 q-tiles = 4096 waves)
  flash_torsion_kernel<<<512, 256, 0, stream>>>(Qb, Kb, Vtb, tor, attn);

  // Output projection -> f32 d_out
  gemm_wmma_kernel<<<ggrid, 256, 0, stream>>>(attn, Wt + 3ull*DMODEL*DMODEL, bo, out, nullptr, 0, 1.0f);
}
